// BuildModel_3796751089795
// MI455X (gfx1250) — compile-verified, matrix operand order checked by reading, not run on hardware
//
#include <hip/hip_runtime.h>

// ============================================================================
// ON-LSTM-ish scan, optimized for MI455X (gfx1250, wave32, WMMA).
//
// Shapes: B=512 T=128 F=256 H=384 L=3 K=10 LAB=25 CH=128 GATES=1542
//
// Design:
//  * PERSISTENT kernel runs all 128 timesteps in one launch. Grid =
//    4 row-groups x 25 n-groups = 100 blocks x 256 threads.
//      - Step-invariant W^T slab for the block's 64 output columns
//        (64 x 672 bf16 ~ 85 KB) is staged into dynamic LDS ONCE before the
//        t-loop (320 KB/WGP LDS makes this possible); B fragments are
//        conflict-free ds_load_b128 (row stride padded 672->680 elems).
//        This removes the ~32x redundant global re-read of W^T every step.
//      - phase A: gate GEMM xo = [x_t|t|h|t] @ [kernel_w;rec_w] + bias via
//        v_wmma_f32_16x16x32_bf16. Wave w of block (rg,ng) computes the
//        16x64 strip (m-tile rg*8+w, cols ng*64..+63). A fragments from
//        global, double-buffered in a #pragma unroll 1 pipelined loop.
//        sched_barrier(0) pins all 8 B-fragment ds_loads BEFORE the 4 WMMAs
//        so DScnt waits are partial, not full drains. Bias is folded into
//        the accumulator init (C/D col = lane&15 -> splat per j).
//      - phase B: cumax gates + ON-LSTM cell; writes h to f32 history AND
//        bf16 A-staging for t+1; stages x_{t+1}; prefetches x_{t+2}.
//    Phases separated by a monotonic-counter grid barrier (zeroed per call
//    -> graph-replay deterministic). Abuf/XO stay L2-resident (192 MB L2).
//  * theme/conv/out tail only matters at t = v_len[b]-1 per row -> computed
//    once at the end from stored h/dist history (saves ~200 GFLOP and most
//    of the serial critical path).
//
// WMMA fragment layout (CDNA5 ISA 7.12.2, 16-bit A 16x32):
//   lane<16 : M=lane,    elems 0..7 -> K=k0..k0+7,   8..15 -> K=k0+16..k0+23
//   lane>=16: M=lane-16, elems 0..7 -> K=k0+8..+15,  8..15 -> K=k0+24..+31
// B (32x16) mirrors A with N=lane when stored N-major (we store W^T).
// C/D: col = lane&15, VGPR r -> row r (+8 for lanes 16..31).
// ============================================================================

#define BB    512
#define TT    128
#define FF    256
#define HH    384
#define LL    3
#define KK    10
#define LABN  25
#define CHN   128          // H / L
#define GATES 1542         // 4H + 2L
#define H6    64           // H / 6
#define KCAT  642          // 257 + 385
#define KPAD  672          // 21 * 32
#define KITER (KPAD / 32)  // 21
#define KLDS  680          // LDS row stride (pad: conflict-free ds_load_b128)
#define NPAD  1600         // 100 * 16  (>= GATES, 4-tile groups all full)
#define NGRP  25           // NPAD / 64
#define NBLK  100          // 4 row-groups * 25 n-groups
#define LDSB_BYTES (64 * KLDS * 2)   // 87,040 B dynamic LDS

typedef __bf16 bf16;
typedef __attribute__((ext_vector_type(16))) __bf16 v16bf;
typedef __attribute__((ext_vector_type(8)))  __bf16 v8bf;
typedef __attribute__((ext_vector_type(8)))  float  v8f;

static __device__ __forceinline__ bf16 f2bf(float f) {
  unsigned u = __builtin_bit_cast(unsigned, f);
  u = (u + 0x7FFFu + ((u >> 16) & 1u)) >> 16;   // round-to-nearest-even
  return __builtin_bit_cast(bf16, (unsigned short)u);
}
static __device__ __forceinline__ float sigm(float x) {
  return 1.0f / (1.0f + __expf(-x));
}
static __device__ __forceinline__ v16bf cat16(v8bf lo, v8bf hi) {
  return __builtin_shufflevector(lo, hi,
           0,1,2,3,4,5,6,7,8,9,10,11,12,13,14,15);
}

// Monotonic-counter grid barrier (blocks assumed co-resident: NBLK=100).
static __device__ __forceinline__ void grid_barrier(unsigned* cnt, unsigned goal) {
  __syncthreads();
  __threadfence();
  if (threadIdx.x == 0) {
    atomicAdd(cnt, 1u);
    while (*((volatile unsigned*)cnt) < goal) __builtin_amdgcn_s_sleep(1);
  }
  __syncthreads();
  __threadfence();
}

// ---------------------------------------------------------------------------
// Build bf16 W^T (N-major) for the concatenated weight [kernel_w ; rec_w],
// zero-padded to NPAD x KPAD, plus combined bias.
// ---------------------------------------------------------------------------
__global__ void prep_weights(const float* __restrict__ kw,
                             const float* __restrict__ kb,
                             const float* __restrict__ rw,
                             const float* __restrict__ rb,
                             bf16* __restrict__ WT,
                             float* __restrict__ biascat) {
  int idx = blockIdx.x * blockDim.x + threadIdx.x;
  if (idx < NPAD * KPAD) {
    int n = idx / KPAD, k = idx % KPAD;
    float v = 0.f;
    if (n < GATES) {
      if (k < 257)        v = kw[(size_t)k * GATES + n];
      else if (k < KCAT)  v = rw[(size_t)(k - 257) * GATES + n];
    }
    WT[idx] = f2bf(v);
  }
  if (idx < NPAD) biascat[idx] = (idx < GATES) ? (kb[idx] + rb[idx]) : 0.f;
}

// ---------------------------------------------------------------------------
// A-staging for t=0: [x_0 | 0 | h=0 | 0 | zero-pad]; also reset barrier cnt.
// ---------------------------------------------------------------------------
__global__ void init_abuf(const float* __restrict__ X, bf16* __restrict__ Abuf,
                          unsigned* __restrict__ barcnt) {
  int idx = blockIdx.x * blockDim.x + threadIdx.x;
  if (idx == 0) *barcnt = 0u;
  if (idx >= BB * KPAD) return;
  int b = idx / KPAD, k = idx % KPAD;
  float v = (k < FF) ? X[(size_t)b * TT * FF + k] : 0.f;   // t = 0
  Abuf[idx] = f2bf(v);
}

// ---------------------------------------------------------------------------
// Persistent scan kernel: 128 steps of (WMMA gate GEMM ; cell update).
// ---------------------------------------------------------------------------
__global__ void __launch_bounds__(256, 1)
scan_persistent(bf16* Abuf,
                const bf16* __restrict__ WT,
                const float* __restrict__ biascat,
                float* __restrict__ XO,
                float* __restrict__ Cbuf,
                float* __restrict__ Hhist,
                float* __restrict__ Dhist,
                const float* __restrict__ X,
                unsigned* __restrict__ barcnt) {
  extern __shared__ bf16 ldsB[];               // 64 cols x KLDS

  const int lane = threadIdx.x & 31;
  const int w    = threadIdx.x >> 5;           // wave in block, 0..7
  const int rg   = blockIdx.x / NGRP;          // row-group 0..3
  const int ng   = blockIdx.x % NGRP;          // n-group   0..24
  const int m0   = (rg * 8 + w) * 16;          // this wave's 16 rows
  const int n0   = ng * 64;                    // block's 64 cols
  const int lrow = lane & 15;
  const int kofs = (lane >> 4) * 8;
  const int hi8  = (lane >> 4) * 8;

  // ---- stage this block's W^T slab into LDS once (step-invariant) --------
  for (int i = threadIdx.x; i < 64 * (KPAD / 8); i += 256) {
    int c  = i / (KPAD / 8);
    int k8 = (i % (KPAD / 8)) * 8;
    *(v8bf*)(ldsB + (size_t)c * KLDS + k8) =
        *(const v8bf*)(WT + (size_t)(n0 + c) * KPAD + k8);
  }
  __syncthreads();

  const bf16* arow = Abuf + (size_t)(m0 + lrow) * KPAD + kofs;
  const bf16* bfrag = ldsB + (size_t)lrow * KLDS + kofs;   // + j*16*KLDS
  float bias[4];
#pragma unroll
  for (int j = 0; j < 4; ++j) bias[j] = biascat[n0 + 16 * j + lrow];

  const int gj  = threadIdx.x & (CHN - 1);   // gate-phase channel
  const int gbo = threadIdx.x >> 7;          // gate-phase row-within-task

  unsigned ep = 0;
  for (int t = 0; t < TT; ++t) {
    // ---------------- phase A: gate GEMM (one 16x64 strip per wave) -------
    {
      v8f acc[4];
#pragma unroll
      for (int j = 0; j < 4; ++j) {
        float bj = bias[j];
        acc[j] = v8f{bj, bj, bj, bj, bj, bj, bj, bj};   // C init = bias
      }
      const bf16* ap = arow;
      v8bf al = *(const v8bf*)(ap);
      v8bf ah = *(const v8bf*)(ap + 16);
      ap += 32;
#pragma unroll 1
      for (int it = 0; it < KITER - 1; ++it) {
        // global loads for iter it+1 overlap the WMMAs of iter it
        v8bf nal = *(const v8bf*)(ap);
        v8bf nah = *(const v8bf*)(ap + 16);
        ap += 32;
        __builtin_amdgcn_sched_barrier(0);   // keep A loads hoisted here
        // issue ALL B-fragment LDS loads before consuming any
        const bf16* lp = bfrag + it * 32;
        v8bf bl[4], bh[4];
#pragma unroll
        for (int j = 0; j < 4; ++j) {
          bl[j] = *(const v8bf*)(lp + (size_t)j * 16 * KLDS);
          bh[j] = *(const v8bf*)(lp + (size_t)j * 16 * KLDS + 16);
        }
        __builtin_amdgcn_sched_barrier(0);   // loads above, WMMAs below
        v16bf a = cat16(al, ah);
#pragma unroll
        for (int j = 0; j < 4; ++j) {
          acc[j] = __builtin_amdgcn_wmma_f32_16x16x32_bf16(
                       false, a, false, cat16(bl[j], bh[j]), (short)0,
                       acc[j], false, false);
        }
        al = nal; ah = nah;
      }
      {  // epilogue k-step
        const bf16* lp = bfrag + (KITER - 1) * 32;
        v8bf bl[4], bh[4];
#pragma unroll
        for (int j = 0; j < 4; ++j) {
          bl[j] = *(const v8bf*)(lp + (size_t)j * 16 * KLDS);
          bh[j] = *(const v8bf*)(lp + (size_t)j * 16 * KLDS + 16);
        }
        __builtin_amdgcn_sched_barrier(0);   // loads above, WMMAs below
        v16bf a = cat16(al, ah);
#pragma unroll
        for (int j = 0; j < 4; ++j) {
          acc[j] = __builtin_amdgcn_wmma_f32_16x16x32_bf16(
                       false, a, false, cat16(bl[j], bh[j]), (short)0,
                       acc[j], false, false);
        }
      }
#pragma unroll
      for (int j = 0; j < 4; ++j) {
        int n = n0 + 16 * j + lrow;
#pragma unroll
        for (int r = 0; r < 8; ++r)
          XO[(size_t)(m0 + r + hi8) * NPAD + n] = acc[j][r];
      }
    }
    ++ep; grid_barrier(barcnt, ep * NBLK);

    // ---------------- phase B: cumax gates + cell update ------------------
    for (int task = blockIdx.x; task < BB / 2; task += NBLK) {
      const int b = 2 * task + gbo;
      const float* xo = XO + (size_t)b * NPAD;

      // cumax l2r over xo[0:3]
      float a0 = xo[0], a1 = xo[1], a2 = xo[2];
      float m = fmaxf(a0, fmaxf(a1, a2));
      float e0 = __expf(a0 - m), e1 = __expf(a1 - m), e2 = __expf(a2 - m);
      float inv = 1.f / (e0 + e1 + e2);
      float fm[LL];
      fm[0] = e0 * inv; fm[1] = fm[0] + e1 * inv; fm[2] = fm[1] + e2 * inv;

      // cumax r2l over xo[3:6]
      float b0 = xo[3], b1 = xo[4], b2 = xo[5];
      m = fmaxf(b0, fmaxf(b1, b2));
      e0 = __expf(b0 - m); e1 = __expf(b1 - m); e2 = __expf(b2 - m);
      inv = 1.f / (e0 + e1 + e2);
      float im[LL];
      im[2] = e2 * inv; im[1] = im[2] + e1 * inv; im[0] = im[1] + e0 * inv;

#pragma unroll
      for (int l = 0; l < LL; ++l) {
        int idx = l * CHN + gj;
        float fg = sigm(xo[2 * LL + (l         ) * CHN + gj]);
        float ig = sigm(xo[2 * LL + (LL + l    ) * CHN + gj]);
        float og = sigm(xo[2 * LL + (2 * LL + l) * CHN + gj]);
        float ci = tanhf(xo[2 * LL + (3 * LL + l) * CHN + gj]);
        float cl = (t == 0) ? 0.f : Cbuf[(size_t)b * HH + idx];
        float ov = fm[l] * im[l];
        float cn = ov * (fg * cl + ig * ci) + (fm[l] - ov) * cl
                 + (im[l] - ov) * ci;
        float hn = og * tanhf(cn);
        Cbuf[(size_t)b * HH + idx] = cn;
        Hhist[((size_t)t * BB + b) * HH + idx] = hn;
        Abuf[(size_t)b * KPAD + 257 + idx] = f2bf(hn);   // h for step t+1
      }

      if (gj == 0)
        Dhist[(size_t)t * BB + b] =
            1.f - (fm[0] + fm[1] + fm[2]) * (1.f / 3.f);

      if (t + 1 < TT) {   // stage x_{t+1}; time flags (t+1 >= 1 -> 1.0)
        const float* xrow = X + ((size_t)b * TT + (t + 1)) * FF;
        for (int f = gj; f < FF; f += CHN)
          Abuf[(size_t)b * KPAD + f] = f2bf(xrow[f]);
        if (gj == 0) {
          Abuf[(size_t)b * KPAD + 256] = f2bf(1.f);
          Abuf[(size_t)b * KPAD + 641] = f2bf(1.f);
        }
      }
      if (t + 2 < TT) {   // prefetch x_{t+2} row into cache hierarchy
        const char* p = (const char*)(X + ((size_t)b * TT + (t + 2)) * FF);
        __builtin_prefetch(p + (size_t)gj * 8, 0, 3);
      }
    }
    ++ep; grid_barrier(barcnt, ep * NBLK);
  }
}

// ---------------------------------------------------------------------------
// Tail: per batch row, at t_sel = v_len-1, rebuild the K=10 window from
// history, compute local_dis, theme MLP, conv einsum, output projection.
// ---------------------------------------------------------------------------
__global__ void __launch_bounds__(256)
finalize(const float* __restrict__ Hhist,
         const float* __restrict__ Dhist,
         const int*   __restrict__ vlen,
         const float* __restrict__ sw,  const float* __restrict__ sb,
         const float* __restrict__ rw2, const float* __restrict__ rb2,
         const float* __restrict__ cw,  const float* __restrict__ cb,
         const float* __restrict__ ow,  const float* __restrict__ ob,
         float* __restrict__ out) {
  __shared__ float ld[KK];
  __shared__ float lhs[HH * KK];   // local_h[h][k]
  __shared__ float th[HH];
  __shared__ float s1[H6];
  __shared__ float rnn[HH];

  const int b = blockIdx.x, tid = threadIdx.x;
  int tsel = vlen[b] - 1;
  tsel = tsel < 0 ? 0 : (tsel >= TT ? TT - 1 : tsel);

  if (tid == 0) {
    float c = 0.f, e[KK], mx = -1e30f;
    for (int k = 0; k < KK; ++k) {
      int ts = tsel - (KK - 1) + k;
      float d = (ts >= 0) ? Dhist[(size_t)ts * BB + b] : 0.f;
      c += d; e[k] = c; mx = fmaxf(mx, c);
    }
    float s = 0.f;
    for (int k = 0; k < KK; ++k) { e[k] = __expf(e[k] - mx); s += e[k]; }
    float is = 1.f / s;
    for (int k = 0; k < KK; ++k) ld[k] = e[k] * is;
  }
  __syncthreads();

  for (int h = tid; h < HH; h += blockDim.x) {
    float acc = 0.f;
#pragma unroll
    for (int k = 0; k < KK; ++k) {
      int ts = tsel - (KK - 1) + k;
      float hv = (ts >= 0) ? Hhist[((size_t)ts * BB + b) * HH + h] : 0.f;
      float v = hv * ld[k];
      lhs[h * KK + k] = v;
      acc += v;
    }
    th[h] = acc * (1.f / KK);
  }
  __syncthreads();

  if (tid < H6) {
    float acc = sb[tid];
    for (int h = 0; h < HH; ++h) acc += th[h] * sw[(size_t)h * H6 + tid];
    s1[tid] = fmaxf(acc, 0.f);
  }
  __syncthreads();

  for (int o = tid; o < HH; o += blockDim.x) {
    float t2 = rb2[o];
    for (int mm = 0; mm < H6; ++mm) t2 += s1[mm] * rw2[(size_t)mm * HH + o];
    t2 = sigm(t2);
    float cv = cb[o];
    const float* cwo = cw + (size_t)o * HH * KK;
    for (int h = 0; h < HH; ++h) {
#pragma unroll
      for (int k = 0; k < KK; ++k) cv += lhs[h * KK + k] * cwo[h * KK + k];
    }
    float hsel = Hhist[((size_t)tsel * BB + b) * HH + o];
    rnn[o] = t2 * cv + hsel;
  }
  __syncthreads();

  if (tid < LABN) {
    float acc = ob[tid];
    for (int h = 0; h < HH; ++h) acc += rnn[h] * ow[(size_t)h * LABN + tid];
    out[(size_t)b * LABN + tid] = acc;
  }
}

// ---------------------------------------------------------------------------
extern "C" void kernel_launch(void* const* d_in, const int* in_sizes, int n_in,
                              void* d_out, int out_size, void* d_ws, size_t ws_size,
                              hipStream_t stream) {
  (void)in_sizes; (void)n_in; (void)out_size; (void)ws_size;
  const float* X   = (const float*)d_in[0];
  const int*   vl  = (const int*)  d_in[1];
  const float* kw  = (const float*)d_in[2];
  const float* kb  = (const float*)d_in[3];
  const float* rw  = (const float*)d_in[4];
  const float* rb  = (const float*)d_in[5];
  const float* sw  = (const float*)d_in[6];
  const float* sb  = (const float*)d_in[7];
  const float* rw2 = (const float*)d_in[8];
  const float* rb2 = (const float*)d_in[9];
  const float* cw  = (const float*)d_in[10];
  const float* cb  = (const float*)d_in[11];
  const float* ow  = (const float*)d_in[12];
  const float* ob  = (const float*)d_in[13];

  char* ws = (char*)d_ws;
  size_t off = 0;
  auto take = [&](size_t bytes) -> char* {
    off = (off + 255) & ~(size_t)255;
    char* p = ws + off;
    off += bytes;
    return p;
  };
  bf16*     WT    = (bf16*)    take((size_t)NPAD * KPAD * sizeof(bf16));
  float*    biasc = (float*)   take((size_t)NPAD * sizeof(float));
  bf16*     Abuf  = (bf16*)    take((size_t)BB * KPAD * sizeof(bf16));
  float*    XO    = (float*)   take((size_t)BB * NPAD * sizeof(float));
  float*    Cbuf  = (float*)   take((size_t)BB * HH * sizeof(float));
  float*    Hh    = (float*)   take((size_t)TT * BB * HH * sizeof(float));
  float*    Dh    = (float*)   take((size_t)TT * BB * sizeof(float));
  unsigned* cnt   = (unsigned*)take(256);

  prep_weights<<<(NPAD * KPAD + 255) / 256, 256, 0, stream>>>(
      kw, kb, rw, rb, WT, biasc);
  init_abuf<<<(BB * KPAD + 255) / 256, 256, 0, stream>>>(X, Abuf, cnt);
  scan_persistent<<<NBLK, 256, LDSB_BYTES, stream>>>(
      Abuf, WT, biasc, XO, Cbuf, Hh, Dh, X, cnt);
  finalize<<<BB, 256, 0, stream>>>(Hh, Dh, vl, sw, sb, rw2, rb2,
                                   cw, cb, ow, ob, (float*)d_out);
}
